// ConvolverBlock_82617990906062
// MI455X (gfx1250) — compile-verified, matrix operand order checked by reading, not run on hardware
//
#include <hip/hip_runtime.h>
#include <math.h>

typedef __attribute__((ext_vector_type(16))) _Float16 v16h;
typedef __attribute__((ext_vector_type(8)))  _Float16 v8h;
typedef __attribute__((ext_vector_type(8)))  float    v8f;

#define DEV __device__ __forceinline__

namespace cvb {
constexpr int BG = 8, NODES = 256, KNN = 16, NG = 12, C = 128, BASIS = 128;
constexpr int M  = BG * NODES;   // 2048 nodes
constexpr int E  = M * KNN;      // 32768 edges
constexpr int RX = E * NG;       // 393216 edge-grid rows
constexpr int RN = M * NG;       // 24576 node-grid rows
constexpr float TWO_PI = 6.283185307179586f;
}

DEV float gelu_tanh(float x) {
  float x3 = x * x * x;
  return 0.5f * x * (1.0f + tanhf(0.7978845608028654f * (x + 0.044715f * x3)));
}

// ---------------- kNN (per-graph, includes self, stable tie order) ----------------
__global__ void cvb_knn(const float* __restrict__ pos, int* __restrict__ srcIdx) {
  int m = blockIdx.x * blockDim.x + threadIdx.x;
  if (m >= cvb::M) return;
  int base = (m >> 8) << 8;
  float px = pos[3*m], py = pos[3*m+1], pz = pos[3*m+2];
  float dist[16]; int idx[16];
  #pragma unroll
  for (int i = 0; i < 16; ++i) { dist[i] = 3.4e38f; idx[i] = 0; }
  for (int j = 0; j < cvb::NODES; ++j) {
    int n = base + j;
    float dx = pos[3*n] - px, dy = pos[3*n+1] - py, dz = pos[3*n+2] - pz;
    float d2 = dx*dx + dy*dy + dz*dz;
    if (d2 < dist[15]) {
      int p = 15;
      while (p > 0 && d2 < dist[p-1]) { dist[p] = dist[p-1]; idx[p] = idx[p-1]; --p; }
      dist[p] = d2; idx[p] = n;
    }
  }
  #pragma unroll
  for (int k = 0; k < 16; ++k) srcIdx[m*16 + k] = idx[k];
}

// ---------------- edge geometry: (xy, z) per (edge, grid point) ----------------
__global__ void cvb_geom(const float* __restrict__ pos, const float* __restrict__ grid,
                         const int* __restrict__ srcIdx, float* __restrict__ geom) {
  int gid = blockIdx.x * blockDim.x + threadIdx.x;
  if (gid >= cvb::E * cvb::NG) return;
  int o = gid % cvb::NG, e = gid / cvb::NG;
  int t = e >> 4;              // target node (edges grouped by target, K contiguous)
  int s = srcIdx[e];
  float rx = pos[3*s]   - pos[3*t];
  float ry = pos[3*s+1] - pos[3*t+1];
  float rz = pos[3*s+2] - pos[3*t+2];
  float gx = grid[3*o], gy = grid[3*o+1], gz = grid[3*o+2];
  float z = rx*gx + ry*gy + rz*gz;
  float dx = rx - z*gx, dy = ry - z*gy, dz = rz - z*gz;
  geom[2*gid]   = sqrtf(dx*dx + dy*dy + dz*dz);
  geom[2*gid+1] = z;
}

// ---------------- RFF features (edge basis input), fp16 out ----------------
__global__ void cvb_rffx(const float* __restrict__ geom, const float* __restrict__ Bx,
                         _Float16* __restrict__ F) {
  int gid = blockIdx.x * blockDim.x + threadIdx.x;
  if (gid >= cvb::RX * 64) return;
  int c2 = gid & 63, row = gid >> 6;
  float xy = geom[2*row], z = geom[2*row+1];
  float proj = cvb::TWO_PI * (xy * Bx[c2] + z * Bx[64 + c2]);
  float sn, cs; __sincosf(proj, &sn, &cs);
  F[(size_t)row*128 + c2]      = (_Float16)cs;
  F[(size_t)row*128 + 64 + c2] = (_Float16)sn;
}

// ---------------- RFF features (rotation basis input: grid@grid^T), fp16 out ----
__global__ void cvb_rffR(const float* __restrict__ grid, const float* __restrict__ BR,
                         _Float16* __restrict__ FR) {
  int gid = blockIdx.x * blockDim.x + threadIdx.x;
  if (gid >= 144 * 64) return;
  int c2 = gid & 63, row = gid >> 6;
  int p = row / cvb::NG, o = row % cvb::NG;
  float a = grid[3*p]*grid[3*o] + grid[3*p+1]*grid[3*o+1] + grid[3*p+2]*grid[3*o+2];
  float proj = cvb::TWO_PI * a * BR[c2];
  float sn, cs; __sincosf(proj, &sn, &cs);
  FR[row*128 + c2]      = (_Float16)cs;
  FR[row*128 + 64 + c2] = (_Float16)sn;
}

// ---------------- weight transpose + f32 -> f16 ----------------
__global__ void cvb_trans(const float* __restrict__ src, _Float16* __restrict__ dst,
                          int K, int N, int total) {
  int gid = blockIdx.x * blockDim.x + threadIdx.x;
  if (gid >= total) return;
  int k = gid % K, n = gid / K;
  dst[(size_t)n*K + k] = (_Float16)src[(size_t)k*N + n];
}

// ---------------- WMMA GEMM: C[M,N] = act(A[M,K] @ BT[N,K]^T + bias) ------------
// 4 waves / block; each wave computes a 16x64 strip (4 accumulators) so the A
// fragment is loaded once per k-chunk and reused across 4 WMMAs (4x less A
// traffic). All 10 fragment loads of a k-chunk are issued BEFORE the 4 WMMAs so
// the compiler can clause them and stage partial s_wait_loadcnt values instead
// of fully draining before every WMMA.
// flags: bit0 = gelu epilogue, bit1 = fp16 output (else fp32).
__global__ void cvb_wmma_gemm(const _Float16* __restrict__ A, const _Float16* __restrict__ BT,
                              const float* __restrict__ bias, void* __restrict__ Cout,
                              int Mr, int N, int K, int flags) {
  int lane = threadIdx.x & 31;
  int wave = threadIdx.x >> 5;
  int tileM = (blockIdx.y * 4 + wave) * 16;
  int tileN = blockIdx.x * 64;
  if (tileM >= Mr) return;                   // wave-uniform: EXEC stays all-ones
  int hl = lane >> 4;                        // which 16-lane half
  int ml = lane & 15;                        // row (A) / column (B) within tile
  const _Float16* arow = A  + (size_t)(tileM + ml) * K + 8 * hl;
  const _Float16* brow = BT + (size_t)(tileN + ml) * K + 8 * hl;
  v8f acc[4] = {};
  union V16 { v16h v; v8h h[2]; };
  for (int kk = 0; kk < K; kk += 32) {
    V16 ua, ub[4];
    // batch all loads first (one clause, pipelined waits)
    ua.h[0] = *(const v8h*)(arow + kk);      // K = kk + 8*hl + [0..7]
    ua.h[1] = *(const v8h*)(arow + kk + 16); // K = kk + 16 + 8*hl + [0..7]
    #pragma unroll
    for (int t = 0; t < 4; ++t) {
      const _Float16* bp = brow + (size_t)t * 16 * K + kk;
      ub[t].h[0] = *(const v8h*)(bp);
      ub[t].h[1] = *(const v8h*)(bp + 16);
    }
    // then consume
    #pragma unroll
    for (int t = 0; t < 4; ++t) {
      acc[t] = __builtin_amdgcn_wmma_f32_16x16x32_f16(false, ua.v, false, ub[t].v,
                                                      (short)0, acc[t], false, false);
    }
  }
  bool do_gelu = (flags & 1) != 0;
  bool out_h16 = (flags & 2) != 0;
  #pragma unroll
  for (int t = 0; t < 4; ++t) {
    int cn = tileN + 16 * t + ml;
    float bs = bias ? bias[cn] : 0.0f;
    #pragma unroll
    for (int r = 0; r < 8; ++r) {
      float v = acc[t][r] + bs;
      if (do_gelu) v = gelu_tanh(v);
      size_t oidx = (size_t)(tileM + r + 8 * hl) * N + cn;
      if (out_h16) ((_Float16*)Cout)[oidx] = (_Float16)v;
      else         ((float*)Cout)[oidx]    = v;
    }
  }
}

// ---------------- gather * kernel, scatter-mean over K=16 contiguous edges ------
__global__ void cvb_aggregate(const _Float16* __restrict__ T, const float* __restrict__ xin,
                              const int* __restrict__ srcIdx, float* __restrict__ x1) {
  int gid = blockIdx.x * blockDim.x + threadIdx.x;
  if (gid >= cvb::RN * cvb::C) return;
  int c = gid & 127, mo = gid >> 7;
  int o = mo % cvb::NG, m = mo / cvb::NG;
  float acc = 0.0f;
  int ebase = m << 4;
  #pragma unroll 4
  for (int k = 0; k < 16; ++k) {
    int e = ebase + k;
    int s = srcIdx[e];
    acc += (float)T[((size_t)e * cvb::NG + o) * 128 + c] *
           xin[((size_t)s * cvb::NG + o) * 128 + c];
  }
  x1[gid] = acc * (1.0f / 16.0f);
}

// ---------------- fiber depthwise conv: x2[m,p,c] = mean_o x1[m,o,c]*kR[p,o,c] ---
__global__ void cvb_fiber(const float* __restrict__ x1, const float* __restrict__ kR,
                          float* __restrict__ x2) {
  int gid = blockIdx.x * blockDim.x + threadIdx.x;
  if (gid >= cvb::RN * cvb::C) return;
  int c = gid & 127, mp = gid >> 7;
  int p = mp % cvb::NG, m = mp / cvb::NG;
  float acc = 0.0f;
  #pragma unroll
  for (int o = 0; o < cvb::NG; ++o)
    acc += x1[(((size_t)m * cvb::NG + o) << 7) + c] * kR[((p * cvb::NG + o) << 7) + c];
  x2[gid] = acc * (1.0f / (float)cvb::NG);
}

// ---------------- LayerNorm over C=128, fp16 out (one wave per row) -------------
__global__ void cvb_layernorm(const float* __restrict__ x2, const float* __restrict__ g,
                              const float* __restrict__ b, _Float16* __restrict__ out) {
  int lane = threadIdx.x & 31;
  int wave = threadIdx.x >> 5;
  int row = blockIdx.x * (blockDim.x >> 5) + wave;
  if (row >= cvb::RN) return;
  const float* r = x2 + (size_t)row * 128;
  float v[4];
  float s = 0.0f;
  #pragma unroll
  for (int i = 0; i < 4; ++i) { v[i] = r[lane + 32*i]; s += v[i]; }
  #pragma unroll
  for (int off = 16; off > 0; off >>= 1) s += __shfl_xor(s, off, 32);
  float mu = s * (1.0f / 128.0f);
  float q = 0.0f;
  #pragma unroll
  for (int i = 0; i < 4; ++i) { float d = v[i] - mu; q += d * d; }
  #pragma unroll
  for (int off = 16; off > 0; off >>= 1) q += __shfl_xor(q, off, 32);
  float inv = rsqrtf(q * (1.0f / 128.0f) + 1e-5f);
  #pragma unroll
  for (int i = 0; i < 4; ++i) {
    int c = lane + 32*i;
    out[(size_t)row * 128 + c] = (_Float16)((v[i] - mu) * inv * g[c] + b[c]);
  }
}

// ---------------- residual with layer scale ----------------
__global__ void cvb_residual(const float* __restrict__ xin, const float* __restrict__ s,
                             const float* __restrict__ h, float* __restrict__ xout) {
  int gid = blockIdx.x * blockDim.x + threadIdx.x;
  if (gid >= cvb::RN * cvb::C) return;
  int c = gid & 127;
  xout[gid] = xin[gid] + s[c] * h[gid];
}

// =====================================================================
extern "C" void kernel_launch(void* const* d_in, const int* in_sizes, int n_in,
                              void* d_out, int out_size, void* d_ws, size_t ws_size,
                              hipStream_t stream) {
  (void)in_sizes; (void)n_in; (void)out_size; (void)ws_size;
  using namespace cvb;
  const float* x0   = (const float*)d_in[0];
  const float* pos  = (const float*)d_in[1];
  const float* grid = (const float*)d_in[2];
  const float* Bx   = (const float*)d_in[3];
  const float* BRw  = (const float*)d_in[4];
  const float* Wx1  = (const float*)d_in[5];  const float* bx1 = (const float*)d_in[6];
  const float* Wx2  = (const float*)d_in[7];  const float* bx2 = (const float*)d_in[8];
  const float* WR1  = (const float*)d_in[9];  const float* bR1 = (const float*)d_in[10];
  const float* WR2  = (const float*)d_in[11]; const float* bR2 = (const float*)d_in[12];
  const float* Wkx[2] = {(const float*)d_in[13], (const float*)d_in[22]};
  const float* WkR[2] = {(const float*)d_in[14], (const float*)d_in[23]};
  const float* lng[2] = {(const float*)d_in[15], (const float*)d_in[24]};
  const float* lnb[2] = {(const float*)d_in[16], (const float*)d_in[25]};
  const float* W1[2]  = {(const float*)d_in[17], (const float*)d_in[26]};
  const float* b1[2]  = {(const float*)d_in[18], (const float*)d_in[27]};
  const float* W2[2]  = {(const float*)d_in[19], (const float*)d_in[28]};
  const float* b2[2]  = {(const float*)d_in[20], (const float*)d_in[29]};
  const float* lsc[2] = {(const float*)d_in[21], (const float*)d_in[30]};

  // ---- carve workspace ----
  char* ws = (char*)d_ws;
  size_t off = 0;
  auto carve = [&](size_t bytes) -> char* {
    char* p = ws + off; off = (off + bytes + 255) & ~(size_t)255; return p;
  };
  int*      srcIdx = (int*)     carve((size_t)E * 4);
  float*    geom   = (float*)   carve((size_t)E * NG * 2 * 4);
  _Float16* BUF0   = (_Float16*)carve((size_t)RX * C * 2);  // F, then KBx
  _Float16* BUF1   = (_Float16*)carve((size_t)RX * C * 2);  // H, then T per layer
  _Float16* FR     = (_Float16*)carve(144 * 128 * 2);
  _Float16* HR     = (_Float16*)carve(144 * 128 * 2);
  _Float16* KBR    = (_Float16*)carve(144 * 128 * 2);
  float*    kRbuf  = (float*)   carve(144 * 128 * 4);
  float*    x1buf  = (float*)   carve((size_t)RN * C * 4);
  float*    x2buf  = (float*)   carve((size_t)RN * C * 4);
  _Float16* hln    = (_Float16*)carve((size_t)RN * C * 2);
  _Float16* G1     = (_Float16*)carve((size_t)RN * 4 * C * 2);
  float*    G2     = (float*)   carve((size_t)RN * C * 4);
  float*    xA     = (float*)   carve((size_t)RN * C * 4);
  _Float16* Wx1T   = (_Float16*)carve(C * C * 2);
  _Float16* Wx2T   = (_Float16*)carve(C * C * 2);
  _Float16* WR1T   = (_Float16*)carve(C * C * 2);
  _Float16* WR2T   = (_Float16*)carve(C * C * 2);
  _Float16* WkxT[2] = {(_Float16*)carve(C*C*2), (_Float16*)carve(C*C*2)};
  _Float16* WkRT[2] = {(_Float16*)carve(C*C*2), (_Float16*)carve(C*C*2)};
  _Float16* W1T[2]  = {(_Float16*)carve(C*4*C*2), (_Float16*)carve(C*4*C*2)};
  _Float16* W2T[2]  = {(_Float16*)carve(4*C*C*2), (_Float16*)carve(4*C*C*2)};

  auto tr = [&](const float* s, _Float16* d, int Kd, int Nd) {
    int tot = Kd * Nd;
    cvb_trans<<<(tot + 255) / 256, 256, 0, stream>>>(s, d, Kd, Nd, tot);
  };
  auto gemm = [&](const _Float16* A, const _Float16* BT, const float* bias, void* Co,
                  int Mr, int Nc, int Kd, int flags) {
    int tilesM = (Mr + 15) / 16;
    dim3 g((unsigned)(Nc / 64), (unsigned)((tilesM + 3) / 4));
    cvb_wmma_gemm<<<g, 128, 0, stream>>>(A, BT, bias, Co, Mr, Nc, Kd, flags);
  };

  // ---- weights: transpose + convert to fp16 ----
  tr(Wx1, Wx1T, C, C); tr(Wx2, Wx2T, C, C);
  tr(WR1, WR1T, C, C); tr(WR2, WR2T, C, C);
  for (int l = 0; l < 2; ++l) {
    tr(Wkx[l], WkxT[l], BASIS, C);
    tr(WkR[l], WkRT[l], BASIS, C);
    tr(W1[l], W1T[l], C, 4 * C);   // -> [512,128]
    tr(W2[l], W2T[l], 4 * C, C);   // -> [128,512]
  }

  // ---- graph / edge features ----
  cvb_knn<<<(M + 255) / 256, 256, 0, stream>>>(pos, srcIdx);
  cvb_geom<<<(E * NG + 255) / 256, 256, 0, stream>>>(pos, grid, srcIdx, geom);
  cvb_rffx<<<(RX * 64 + 255) / 256, 256, 0, stream>>>(geom, Bx, BUF0);
  cvb_rffR<<<(144 * 64 + 255) / 256, 256, 0, stream>>>(grid, BRw, FR);

  // ---- basis MLPs (WMMA): kb_x = gelu(gelu(F@Wx1+b)@Wx2+b), same for kb_R ----
  gemm(BUF0, Wx1T, bx1, BUF1, RX, C, C, /*gelu+f16*/3);   // H
  gemm(BUF1, Wx2T, bx2, BUF0, RX, C, C, 3);               // KBx (overwrites F)
  gemm(FR,   WR1T, bR1, HR, 144, C, C, 3);
  gemm(HR,   WR2T, bR2, KBR, 144, C, C, 3);

  // ---- two interaction layers ----
  const float* xin = x0;
  for (int l = 0; l < 2; ++l) {
    float* xout = (l == 0) ? xA : (float*)d_out;
    // kR = kb_R @ WkR  [144,128] fp32
    gemm(KBR, WkRT[l], nullptr, kRbuf, 144, C, C, 0);
    // T = kb_x @ Wkx  [RX,128] fp16
    gemm(BUF0, WkxT[l], nullptr, BUF1, RX, C, C, /*f16*/2);
    // x1 = scatter-mean( T * x[src] )
    cvb_aggregate<<<(RN * C + 255) / 256, 256, 0, stream>>>(BUF1, xin, srcIdx, x1buf);
    // x2 = fiber depthwise conv
    cvb_fiber<<<(RN * C + 255) / 256, 256, 0, stream>>>(x1buf, kRbuf, x2buf);
    // ConvNeXt MLP: LN -> 128->512 gelu -> 512->128, layer-scale residual
    cvb_layernorm<<<RN / 8, 256, 0, stream>>>(x2buf, lng[l], lnb[l], hln);
    gemm(hln, W1T[l], b1[l], G1, RN, 4 * C, C, 3);
    gemm(G1, W2T[l], b2[l], G2, RN, C, 4 * C, 0);
    cvb_residual<<<(RN * C + 255) / 256, 256, 0, stream>>>(xin, lsc[l], G2, xout);
    xin = xout;
  }
}
